// MultHeadAtten_8435315769410
// MI455X (gfx1250) — compile-verified, matrix operand order checked by reading, not run on hardware
//
#include <hip/hip_runtime.h>

// ---------------------------------------------------------------------------
// Multi-head attention (B=1, S=4096, D=768, H=12, Dk=64) for MI455X / gfx1250.
// All matmuls run on v_wmma_f32_16x16x32_bf16 (wave32), fp32 accumulation,
// flash-attention online softmax in fp32.  Attention K/V^T tiles are staged
// into LDS by the Tensor Data Mover (tensor_load_to_lds, TENSORcnt), double
// buffered so the DMA of tile kb+1 overlaps the WMMA compute of tile kb.
// ---------------------------------------------------------------------------

#define S_LEN 4096
#define DM    768
#define NH    12
#define DK    64
#define NKB   (S_LEN / 64)

typedef __bf16 bf16_t;
typedef unsigned int u32;
typedef __attribute__((ext_vector_type(16))) __bf16 v16bf;
typedef __attribute__((ext_vector_type(8)))  __bf16 v8bf;
typedef __attribute__((ext_vector_type(4)))  __bf16 v4bf;
typedef __attribute__((ext_vector_type(8)))  float  v8f;
typedef __attribute__((ext_vector_type(4)))  float  v4f;
typedef __attribute__((ext_vector_type(4)))  u32    v4u32;
typedef __attribute__((ext_vector_type(8)))  int    v8i32;
typedef __attribute__((ext_vector_type(4)))  int    v4i32;

__device__ __forceinline__ v8f wmma_bf16(v16bf a, v16bf b, v8f c) {
  // (neg_a, A, neg_b, B, c_mod, C, reuse_a, reuse_b)
  return __builtin_amdgcn_wmma_f32_16x16x32_bf16(false, a, false, b,
                                                 (short)0, c, false, false);
}

// Load one 16x32 (rows x K) bf16 operand fragment from a row-major buffer.
// ISA layout (16-bit A, wave32): lanes 0-15 hold row=lane, K={0..7,16..23};
// lanes 16-31 hold row=lane-16, K={8..15,24..31}. B operands use the same
// per-lane pattern with "row" meaning the N-column of B (i.e. a row of B^T).
__device__ __forceinline__ v16bf load_frag16(const bf16_t* base, int stride) {
  const int lane = threadIdx.x & 31;
  const int r    = lane & 15;
  const int hi   = lane >> 4;
  const bf16_t* p = base + (size_t)r * stride + hi * 8;
  v8bf lo = *(const v8bf*)(p);        // K = hi*8 .. hi*8+7
  v8bf hh = *(const v8bf*)(p + 16);   // K = 16 + hi*8 ..
  v16bf f;
#pragma unroll
  for (int i = 0; i < 8; ++i) { f[i] = lo[i]; f[i + 8] = hh[i]; }
  return f;
}

// ---------------------------------------------------------------------------
// Tensor Data Mover: DMA one 64x64 bf16 tile (row stride `row_stride_elems`
// in global memory) into LDS at byte offset `lds_byte_off`, inserting 16
// bytes of padding after every 128-byte row so LDS rows are KSTR=72 bf16.
// D# layout per CDNA5 ISA:
//   G0: [1:0]=count=1, [63:32]=lds_addr, [120:64]=global_addr, [127:126]=2
//   G1: [17:16]=data_size(1->2B), [20]=pad_enable, [24:22]=pad_interval
//       (4 -> 32 DWORDs), [31:25]=pad_amount (3 -> 4 DWORDs),
//       tensor_dim0/1 (kept huge: tile is known in-bounds),
//       [127:112]=tile_dim0=64, [143:128]=tile_dim1=64,
//       [207:160]=tensor_dim0_stride.
// Wave-level op (EXEC ignored); tracked with TENSORcnt.
// ---------------------------------------------------------------------------
__device__ __forceinline__ void tdm_load_tile64x64(u32 lds_byte_off,
                                                   const bf16_t* gsrc,
                                                   u32 row_stride_elems) {
  const unsigned long long ga = (unsigned long long)(size_t)gsrc;
  const u32 dim0 = 1u << 20, dim1 = 1u << 20;   // generous OOB bounds
  v4u32 g0;
  g0[0] = 1u;                                   // count=1 (valid, user D#)
  g0[1] = lds_byte_off;                         // lds_addr
  g0[2] = (u32)ga;                              // global_addr[31:0]
  g0[3] = (u32)(ga >> 32) | (2u << 30);         // global_addr[56:32]|type=2
  v8i32 g1;
  g1[0] = (int)((1u << 16) | (1u << 20) | (4u << 22) | (3u << 25));
  g1[1] = (int)((dim0 & 0xFFFFu) << 16);        // tensor_dim0[15:0]
  g1[2] = (int)((dim0 >> 16) | ((dim1 & 0xFFFFu) << 16));
  g1[3] = (int)((dim1 >> 16) | (64u << 16));    // tile_dim0 = 64
  g1[4] = 64;                                   // tile_dim1=64, tile_dim2=0
  g1[5] = (int)row_stride_elems;                // tensor_dim0_stride[31:0]
  g1[6] = 0;                                    // stride0[47:32], stride1 lo
  g1[7] = 0;
  v4i32 gz = {0, 0, 0, 0};
#if defined(__clang_major__) && (__clang_major__ >= 23)
  v8i32 gz8 = {0, 0, 0, 0, 0, 0, 0, 0};
  __builtin_amdgcn_tensor_load_to_lds(g0, g1, gz, gz, gz8, 0);
#else
  __builtin_amdgcn_tensor_load_to_lds(g0, g1, gz, gz, 0);
#endif
}

// Generic LDS pointer -> 32-bit LDS byte offset (aperture in high 32 bits).
__device__ __forceinline__ u32 lds_off(const void* p) {
  return (u32)(size_t)p;
}

// ---------------------------------------------------------------------------
// Kernel 1: fused QKV projection.  Y = X @ Wq^T + bq  for X in {q,k,v}
// (the reference applies Wq to all three).  Output per head, bf16:
//   z=0 -> Qh[h][s][d], z=1 -> Kh[h][s][d], z=2 -> VT[h][d][s] (transposed).
// Block: 256 threads = 8 waves; tile 128 rows x 64 cols (one head band);
// K staged in chunks of 32 through LDS with fp32->bf16 conversion.
// ---------------------------------------------------------------------------
#define PSTR 40   // padded LDS row stride (bf16 elems) for 32-wide K chunks

__global__ __launch_bounds__(256)
void qkv_proj_kernel(const float* __restrict__ q,
                     const float* __restrict__ k,
                     const float* __restrict__ v,
                     const float* __restrict__ Wq,
                     const float* __restrict__ bq,
                     bf16_t* __restrict__ Qh,
                     bf16_t* __restrict__ Kh,
                     bf16_t* __restrict__ VT)
{
  __shared__ bf16_t Xs[128][PSTR];
  __shared__ bf16_t Ws[64][PSTR];

  const int tid  = threadIdx.x;
  const int lane = tid & 31;
  const int wave = tid >> 5;           // 0..7
  const int wm   = wave & 3;           // 32-row quadrant
  const int wn   = wave >> 2;          // 32-col half
  const int row0 = blockIdx.x * 128;
  const int h    = blockIdx.y;         // 64-col band == head
  const int col0 = h * DK;
  const int z    = blockIdx.z;         // 0:q 1:k 2:v
  const float* X = (z == 0) ? q : (z == 1) ? k : v;

  v8f acc[2][2];
#pragma unroll
  for (int mi = 0; mi < 2; ++mi)
#pragma unroll
    for (int ni = 0; ni < 2; ++ni) acc[mi][ni] = 0.f;

  for (int k0 = 0; k0 < DM; k0 += 32) {
    __syncthreads();
    // X tile: 128 rows x 32 cols fp32 -> bf16 LDS (1024 float4 / 256 thr)
#pragma unroll
    for (int i = 0; i < 4; ++i) {
      int c  = tid + i * 256;
      int rr = c >> 3, c4 = (c & 7) * 4;
      v4f xv = *(const v4f*)(X + (size_t)(row0 + rr) * DM + k0 + c4);
      v4bf xb;
#pragma unroll
      for (int j = 0; j < 4; ++j) xb[j] = (bf16_t)xv[j];
      *(v4bf*)&Xs[rr][c4] = xb;
    }
    // W tile: 64 rows (d_out) x 32 cols fp32 -> bf16 LDS
#pragma unroll
    for (int i = 0; i < 2; ++i) {
      int c  = tid + i * 256;
      int rr = c >> 3, c4 = (c & 7) * 4;
      v4f wv = *(const v4f*)(Wq + (size_t)(col0 + rr) * DM + k0 + c4);
      v4bf wb;
#pragma unroll
      for (int j = 0; j < 4; ++j) wb[j] = (bf16_t)wv[j];
      *(v4bf*)&Ws[rr][c4] = wb;
    }
    if (k0 + 32 < DM) {   // global_prefetch_b8 for next K-chunk
      __builtin_prefetch(X + (size_t)(row0 + (tid >> 1)) * DM + k0 + 32 +
                             (tid & 1) * 16, 0, 1);
      __builtin_prefetch(Wq + (size_t)(col0 + (tid & 63)) * DM + k0 + 32,
                         0, 1);
    }
    __syncthreads();

#pragma unroll
    for (int mi = 0; mi < 2; ++mi) {
      v16bf a = load_frag16(&Xs[wm * 32 + mi * 16][0], PSTR);
#pragma unroll
      for (int ni = 0; ni < 2; ++ni) {
        v16bf b = load_frag16(&Ws[wn * 32 + ni * 16][0], PSTR);
        acc[mi][ni] = wmma_bf16(a, b, acc[mi][ni]);
      }
    }
  }

  // Epilogue: bias + store.  C/D layout: lane holds column n; VGPR r holds
  // row m = r + 8*(lane>=16).
  const int hi = lane >> 4;
  const int nl = lane & 15;
#pragma unroll
  for (int mi = 0; mi < 2; ++mi) {
#pragma unroll
    for (int ni = 0; ni < 2; ++ni) {
      const int   d    = wn * 32 + ni * 16 + nl;
      const float bias = bq[col0 + d];
      const int   s0   = row0 + wm * 32 + mi * 16 + hi * 8;
      if (z < 2) {
        bf16_t* O = (z == 0) ? Qh : Kh;
        bf16_t* p = O + ((size_t)h * S_LEN + s0) * DK + d;
#pragma unroll
        for (int r = 0; r < 8; ++r)
          p[(size_t)r * DK] = (bf16_t)(acc[mi][ni][r] + bias);
      } else {
        // transposed store: lane's 8 rows are 8 consecutive s -> one 16B store
        v8bf pv;
#pragma unroll
        for (int r = 0; r < 8; ++r) pv[r] = (bf16_t)(acc[mi][ni][r] + bias);
        *(v8bf*)(VT + (size_t)(h * DK + d) * S_LEN + s0) = pv;
      }
    }
  }
}

// ---------------------------------------------------------------------------
// Kernel 2: flash attention per head.  Block = 128 threads = 4 waves; each
// wave owns 16 query rows.  64-key tiles of K (row-major) and V^T are DMA'd
// into double-buffered LDS by the TDM (wave 0 issues, all waves consume after
// a barrier); online softmax in fp32 with 16-lane shfl reductions (matches
// C/D lane layout); P goes through per-wave LDS to reach A-operand layout.
// ---------------------------------------------------------------------------
#define KSTR 72   // padded LDS row stride (bf16), 144B rows (16B aligned)

__global__ __launch_bounds__(128)
void attention_kernel(const bf16_t* __restrict__ Qh,
                      const bf16_t* __restrict__ Kh,
                      const bf16_t* __restrict__ VT,
                      bf16_t* __restrict__ AO)
{
  __shared__ bf16_t Ks[2][64][KSTR];   // keys (rows=key, cols=dk), 2 buffers
  __shared__ bf16_t Vs[2][64][KSTR];   // V^T  (rows=d,   cols=key), 2 buffers
  __shared__ bf16_t Ps[4][16][KSTR];   // per-wave P tile (rows=q, cols=key)

  const int tid  = threadIdx.x;
  const int lane = tid & 31;
  const int wave = tid >> 5;                 // 0..3
  const int h    = blockIdx.y;
  const int q0   = blockIdx.x * 64 + wave * 16;
  const int hi   = lane >> 4;
  const int nl   = lane & 15;

  const bf16_t* kgbase = Kh + (size_t)h * S_LEN * DK;   // [s][dk]
  const bf16_t* vgbase = VT + (size_t)h * DK * S_LEN;   // [d][s]

  // Q fragments kept in registers for the whole sweep (A operand, dk=64)
  const bf16_t* qbase = Qh + ((size_t)h * S_LEN + q0) * DK;
  v16bf qf0 = load_frag16(qbase +  0, DK);
  v16bf qf1 = load_frag16(qbase + 32, DK);

  v8f o[4];
#pragma unroll
  for (int ni = 0; ni < 4; ++ni) o[ni] = 0.f;
  float m_s[8], l_s[8];
#pragma unroll
  for (int r = 0; r < 8; ++r) { m_s[r] = -3.0e38f; l_s[r] = 0.f; }

  const float sscale = 0.125f;          // 1/sqrt(64)
  const float log2e  = 1.44269504089f;

  // Prologue: TDM-load tile 0 into buffer 0.
  if (wave == 0) {
    tdm_load_tile64x64(lds_off(&Ks[0][0][0]), kgbase, DK);
    tdm_load_tile64x64(lds_off(&Vs[0][0][0]), vgbase, S_LEN);
    __builtin_amdgcn_s_wait_tensorcnt(0);
  }
  __syncthreads();

  for (int kb = 0; kb < NKB; ++kb) {
    const int buf = kb & 1;

    // Kick off the DMA for the next tile into the other buffer; it runs
    // concurrently with this iteration's WMMA/softmax work.
    if (wave == 0 && (kb + 1) < NKB) {
      tdm_load_tile64x64(lds_off(&Ks[buf ^ 1][0][0]),
                         kgbase + (size_t)(kb + 1) * 64 * DK, DK);
      tdm_load_tile64x64(lds_off(&Vs[buf ^ 1][0][0]),
                         vgbase + (kb + 1) * 64, S_LEN);
    }

    // S = Q K^T  (16 x 64), fp32 accum: 8 WMMAs
    v8f sa[4];
#pragma unroll
    for (int ni = 0; ni < 4; ++ni) sa[ni] = 0.f;
#pragma unroll
    for (int ni = 0; ni < 4; ++ni) {
      v16bf b0 = load_frag16(&Ks[buf][ni * 16][0],  KSTR);
      v16bf b1 = load_frag16(&Ks[buf][ni * 16][32], KSTR);
      sa[ni] = wmma_bf16(qf0, b0, sa[ni]);
      sa[ni] = wmma_bf16(qf1, b1, sa[ni]);
    }

    // Online softmax.  Row m = r + 8*hi lives across one 16-lane half.
#pragma unroll
    for (int r = 0; r < 8; ++r) {
      float mx = -3.0e38f;
#pragma unroll
      for (int ni = 0; ni < 4; ++ni) {
        sa[ni][r] *= sscale;
        mx = fmaxf(mx, sa[ni][r]);
      }
#pragma unroll
      for (int off = 8; off > 0; off >>= 1)
        mx = fmaxf(mx, __shfl_xor(mx, off, 32));
      const float mnew = fmaxf(m_s[r], mx);
      const float corr = exp2f((m_s[r] - mnew) * log2e);
      m_s[r] = mnew;
      float rs = 0.f;
#pragma unroll
      for (int ni = 0; ni < 4; ++ni) {
        float pv = exp2f((sa[ni][r] - mnew) * log2e);
        sa[ni][r] = pv;                 // reuse score regs for P
        rs += pv;
      }
#pragma unroll
      for (int off = 8; off > 0; off >>= 1)
        rs += __shfl_xor(rs, off, 32);
      l_s[r] = l_s[r] * corr + rs;
#pragma unroll
      for (int ni = 0; ni < 4; ++ni) o[ni][r] *= corr;
    }

    // P (C/D layout) -> per-wave LDS row-major, then reload as A operand.
#pragma unroll
    for (int ni = 0; ni < 4; ++ni)
#pragma unroll
      for (int r = 0; r < 8; ++r)
        Ps[wave][r + hi * 8][ni * 16 + nl] = (bf16_t)sa[ni][r];
    __builtin_amdgcn_wave_barrier();    // keep same-wave DS ops ordered

    // O += P V : A = P (16x64) from LDS, B = V^T rows from Vs: 8 WMMAs
#pragma unroll
    for (int c = 0; c < 2; ++c) {
      v16bf a = load_frag16(&Ps[wave][0][c * 32], KSTR);
#pragma unroll
      for (int ni = 0; ni < 4; ++ni) {
        v16bf b = load_frag16(&Vs[buf][ni * 16][c * 32], KSTR);
        o[ni] = wmma_bf16(a, b, o[ni]);
      }
    }

    // Drain the in-flight DMA before everyone flips to the other buffer.
    if (wave == 0 && (kb + 1) < NKB)
      __builtin_amdgcn_s_wait_tensorcnt(0);
    __syncthreads();
  }

  // Normalize and store concat layout AO[s][h*64+d] (bf16)
#pragma unroll
  for (int ni = 0; ni < 4; ++ni) {
    const int d = ni * 16 + nl;
    bf16_t* p = AO + (size_t)(q0 + hi * 8) * DM + h * DK + d;
#pragma unroll
    for (int r = 0; r < 8; ++r)
      p[(size_t)r * DM] = (bf16_t)(o[ni][r] / l_s[r]);
  }
}

// ---------------------------------------------------------------------------
// Kernel 3: output projection.  out = AO @ Wo^T + bo  (bf16 x bf16 -> fp32)
// Same tiling as kernel 1; A is already bf16, W converted on load.
// ---------------------------------------------------------------------------
__global__ __launch_bounds__(256)
void out_proj_kernel(const bf16_t* __restrict__ AO,
                     const float* __restrict__ Wo,
                     const float* __restrict__ bo,
                     float* __restrict__ Y)
{
  __shared__ bf16_t As[128][PSTR];
  __shared__ bf16_t Ws[64][PSTR];

  const int tid  = threadIdx.x;
  const int lane = tid & 31;
  const int wave = tid >> 5;
  const int wm   = wave & 3;
  const int wn   = wave >> 2;
  const int row0 = blockIdx.x * 128;
  const int col0 = blockIdx.y * 64;

  v8f acc[2][2];
#pragma unroll
  for (int mi = 0; mi < 2; ++mi)
#pragma unroll
    for (int ni = 0; ni < 2; ++ni) acc[mi][ni] = 0.f;

  for (int k0 = 0; k0 < DM; k0 += 32) {
    __syncthreads();
    // A tile: 128 x 32 bf16 (512 x 16B chunks / 256 threads)
#pragma unroll
    for (int i = 0; i < 2; ++i) {
      int c  = tid + i * 256;
      int rr = c >> 2, c8 = (c & 3) * 8;
      *(v8bf*)&As[rr][c8] =
          *(const v8bf*)(AO + (size_t)(row0 + rr) * DM + k0 + c8);
    }
    // W tile: 64 x 32 fp32 -> bf16
#pragma unroll
    for (int i = 0; i < 2; ++i) {
      int c  = tid + i * 256;
      int rr = c >> 3, c4 = (c & 7) * 4;
      v4f wv = *(const v4f*)(Wo + (size_t)(col0 + rr) * DM + k0 + c4);
      v4bf wb;
#pragma unroll
      for (int j = 0; j < 4; ++j) wb[j] = (bf16_t)wv[j];
      *(v4bf*)&Ws[rr][c4] = wb;
    }
    if (k0 + 32 < DM) {   // global_prefetch_b8 for next K-chunk
      __builtin_prefetch(AO + (size_t)(row0 + (tid >> 1)) * DM + k0 + 32,
                         0, 1);
      __builtin_prefetch(Wo + (size_t)(col0 + (tid & 63)) * DM + k0 + 32,
                         0, 1);
    }
    __syncthreads();

#pragma unroll
    for (int mi = 0; mi < 2; ++mi) {
      v16bf a = load_frag16(&As[wm * 32 + mi * 16][0], PSTR);
#pragma unroll
      for (int ni = 0; ni < 2; ++ni) {
        v16bf b = load_frag16(&Ws[wn * 32 + ni * 16][0], PSTR);
        acc[mi][ni] = wmma_bf16(a, b, acc[mi][ni]);
      }
    }
  }

  const int hi = lane >> 4;
  const int nl = lane & 15;
#pragma unroll
  for (int mi = 0; mi < 2; ++mi) {
#pragma unroll
    for (int ni = 0; ni < 2; ++ni) {
      const int   cg   = col0 + wn * 32 + ni * 16 + nl;
      const float bias = bo[cg];
      float* p = Y + (size_t)(row0 + wm * 32 + mi * 16 + hi * 8) * DM + cg;
#pragma unroll
      for (int r = 0; r < 8; ++r)
        p[(size_t)r * DM] = acc[mi][ni][r] + bias;
    }
  }
}

// ---------------------------------------------------------------------------
extern "C" void kernel_launch(void* const* d_in, const int* in_sizes, int n_in,
                              void* d_out, int out_size, void* d_ws,
                              size_t ws_size, hipStream_t stream) {
  (void)in_sizes; (void)n_in; (void)out_size;

  const float* q  = (const float*)d_in[0];
  const float* k  = (const float*)d_in[1];
  const float* v  = (const float*)d_in[2];
  const float* Wq = (const float*)d_in[3];
  const float* bq = (const float*)d_in[4];
  const float* Wo = (const float*)d_in[5];
  const float* bo = (const float*)d_in[6];
  float* out = (float*)d_out;

  // Workspace layout (bf16): Qh | Kh | VT | AO ; 4 * 6.29 MB = 25.2 MB
  bf16_t* wsb = (bf16_t*)d_ws;
  const size_t qe = (size_t)NH * S_LEN * DK;   // 3,145,728 elems per buffer
  bf16_t* Qh = wsb;
  bf16_t* Kh = wsb + qe;
  bf16_t* VT = wsb + 2 * qe;
  bf16_t* AO = wsb + 3 * qe;
  (void)ws_size;  // requires >= 4 * qe * sizeof(bf16) = 25,165,824 bytes

  qkv_proj_kernel<<<dim3(S_LEN / 128, NH, 3), 256, 0, stream>>>(
      q, k, v, Wq, bq, Qh, Kh, VT);
  attention_kernel<<<dim3(S_LEN / 64, NH), 128, 0, stream>>>(Qh, Kh, VT, AO);
  out_proj_kernel<<<dim3(S_LEN / 128, DM / 64), 256, 0, stream>>>(
      AO, Wo, bo, out);
}
